// GumbelSoftmaxQuantizer_90151363543496
// MI455X (gfx1250) — matrix operand order repaired and checked by reading, not compiled, not run on hardware
//
#include <hip/hip_runtime.h>

// ---------------------------------------------------------------------------
// GumbelSoftmaxQuantizer forward for MI455X (gfx1250, wave32, WMMA)
//   z:[64,64,32,32] f32, u:[64,64,1024] f32, embedding:[64,1024,1024] f32
//   out = [ q_st (4194304 f32) | commit_loss (1 f32) | indices (4096 f32) ]
// ---------------------------------------------------------------------------

#define B_   64
#define C_   64
#define D_   1024
#define K_   1024
#define KC   64      // k-chunk staged in LDS per iteration
#define NT   128     // N (codebook) columns per workgroup
#define NTILES (K_ / NT)   // 8
#define EPS_ 0.005f

typedef __attribute__((ext_vector_type(16))) __bf16 v16bf;
typedef __attribute__((ext_vector_type(8)))  float  v8f;

union BF16Frag {
    unsigned short u[16];
    v16bf v;
};

static __device__ __forceinline__ unsigned short f2bf(float f) {
    unsigned int x = __float_as_uint(f);
    unsigned int r = x + 0x7FFFu + ((x >> 16) & 1u);   // RNE
    return (unsigned short)(r >> 16);
}
static __device__ __forceinline__ unsigned int pack2bf(float lo, float hi) {
    return (unsigned int)f2bf(lo) | ((unsigned int)f2bf(hi) << 16);
}
static __device__ __forceinline__ float bf2f(unsigned short s) {
    return __uint_as_float(((unsigned int)s) << 16);
}

// ---------------------------------------------------------------------------
// Kernel 1: per-channel GEMM xe = Z * E^T via v_wmma_f32_16x16x32_bf16,
// fused ||z||^2 / ||e||^2 reductions (embedding streamed from HBM once),
// fused clip + Gumbel noise; writes score[b][c][k] to scratch.
// grid = C_*NTILES blocks, 256 threads (8 waves), each wave owns a 64x16 strip.
// ---------------------------------------------------------------------------
__global__ __launch_bounds__(256) void gsq_gemm_score(
    const float* __restrict__ z,
    const float* __restrict__ u,
    const float* __restrict__ emb,
    float* __restrict__ scores)
{
    __shared__ __align__(16) unsigned short Zs[B_ * KC];   // [b][d]   8 KB bf16
    __shared__ __align__(16) unsigned short Es[NT * KC];   // [n][d]  16 KB bf16
    __shared__ float x2s[B_];
    __shared__ float e2s[NT];

    const int c    = blockIdx.x >> 3;        // channel
    const int n0   = (blockIdx.x & 7) * NT;  // codebook column base
    const int t    = threadIdx.x;
    const int lane = t & 31;
    const int wave = t >> 5;                 // 0..7 -> 16-col sub-strip
    const int hl   = lane >> 4;              // half-wave selector
    const int l16  = lane & 15;

    v8f acc[4] = {};                         // 4 M-tiles (rows 0..63) x 16 cols
    float x2p = 0.0f, e2p = 0.0f;

    for (int kc = 0; kc < D_; kc += KC) {
        // speculative prefetch of next chunk's embedding strip (global_prefetch_b8)
        if (kc + KC < D_) {
            const char* p = (const char*)(emb +
                (size_t)(c * K_ + n0 + (t >> 1)) * D_ + kc + KC + (t & 1) * 32);
            __builtin_prefetch(p, 0, 1);
        }
        __syncthreads();                     // previous chunk fully consumed
        // ---- stage Z chunk (64x64 f32 -> bf16): float4 loads, b64 LDS stores ----
        #pragma unroll
        for (int j = 0; j < 4; ++j) {
            int e = j * 256 + t;             // 0..1023 float4 slots
            int b = e >> 4, d4 = e & 15;
            float4 v = *(const float4*)(z + (size_t)(b * C_ + c) * D_ + kc + d4 * 4);
            uint2 pk; pk.x = pack2bf(v.x, v.y); pk.y = pack2bf(v.z, v.w);
            *(uint2*)&Zs[b * KC + d4 * 4] = pk;
        }
        // ---- stage E chunk (128x64 f32 -> bf16): float4 loads, b64 LDS stores ----
        #pragma unroll
        for (int j = 0; j < 8; ++j) {
            int e = j * 256 + t;             // 0..2047 float4 slots
            int n = e >> 4, d4 = e & 15;
            float4 v = *(const float4*)(emb + (size_t)(c * K_ + n0 + n) * D_ + kc + d4 * 4);
            uint2 pk; pk.x = pack2bf(v.x, v.y); pk.y = pack2bf(v.z, v.w);
            *(uint2*)&Es[n * KC + d4 * 4] = pk;
        }
        __syncthreads();

        // ---- fused row-norm partials from the staged tiles ----
        if (t < B_) {
            #pragma unroll 8
            for (int d = 0; d < KC; ++d) { float v = bf2f(Zs[t * KC + d]); x2p += v * v; }
        } else if (t < B_ + NT) {
            int n = t - B_;
            #pragma unroll 8
            for (int d = 0; d < KC; ++d) { float v = bf2f(Es[n * KC + d]); e2p += v * v; }
        }

        // ---- WMMA: 2 k-steps of 32 per chunk, 4 M-tiles per wave ----
        #pragma unroll
        for (int ks = 0; ks < KC; ks += 32) {
            // B fragment: 32x16 bf16. N=lane%16; lane half picks K blocks of 8
            //             (same K order as A fragment)  (ISA 7.12.2/7.12.4)
            BF16Frag bf;
            #pragma unroll
            for (int j = 0; j < 16; ++j)
                bf.u[j] = Es[(wave * 16 + l16) * KC + ks + (j >> 3) * 16 + hl * 8 + (j & 7)];

            #pragma unroll
            for (int mt = 0; mt < 4; ++mt) {
                // A fragment: 16x32 bf16. lane half picks K blocks of 8 (ISA 7.12.2)
                BF16Frag af;
                const int m = mt * 16 + l16;
                #pragma unroll
                for (int j = 0; j < 8; ++j) {
                    af.u[j]     = Zs[m * KC + ks + hl * 8 + j];
                    af.u[j + 8] = Zs[m * KC + ks + 16 + hl * 8 + j];
                }
                acc[mt] = __builtin_amdgcn_wmma_f32_16x16x32_bf16(
                    false, af.v, false, bf.v, (short)0, acc[mt], false, false);
            }
        }
    }

    if (t < B_)            x2s[t]      = x2p;
    else if (t < B_ + NT)  e2s[t - B_] = e2p;
    __syncthreads();

    // ---- epilogue: score = -clip(x2 - 2*xe + e2, -5, 5) + gumbel(u) ----
    const int nloc = wave * 16 + l16;       // 0..127
    const int kk   = n0 + nloc;             // global codebook index
    const float e2 = e2s[nloc];
    #pragma unroll
    for (int mt = 0; mt < 4; ++mt) {
        #pragma unroll
        for (int v = 0; v < 8; ++v) {
            const int m = mt * 16 + hl * 8 + v;     // batch row (C/D layout)
            float dist = x2s[m] - 2.0f * acc[mt][v] + e2;
            dist = fminf(5.0f, fmaxf(-5.0f, dist));
            float uv = u[(size_t)(m * C_ + c) * K_ + kk];
            uv = fminf(1.0f - EPS_, fmaxf(EPS_, uv));
            float g = -logf(-logf(uv));             // f32 Gumbel, matches ref
            scores[(size_t)(m * C_ + c) * K_ + kk] = -dist + g;
        }
    }
}

// ---------------------------------------------------------------------------
// Kernel 2: argmax over K=1024 per (b,c); first-index tie-break (jnp.argmax).
// ---------------------------------------------------------------------------
__global__ __launch_bounds__(256) void gsq_argmax(
    const float* __restrict__ scores,
    int* __restrict__ idxbuf,
    float* __restrict__ out_idx)
{
    __shared__ float sv[256];
    __shared__ int   si[256];
    const int bc = blockIdx.x;
    const int t  = threadIdx.x;
    const float4* row = (const float4*)(scores + (size_t)bc * K_);

    float4 v = row[t];                      // k = 4t .. 4t+3, ascending
    float best = v.x; int bi = 4 * t;
    if (v.y > best) { best = v.y; bi = 4 * t + 1; }
    if (v.z > best) { best = v.z; bi = 4 * t + 2; }
    if (v.w > best) { best = v.w; bi = 4 * t + 3; }

    sv[t] = best; si[t] = bi;
    __syncthreads();
    for (int s = 128; s > 0; s >>= 1) {
        if (t < s) {
            float ov = sv[t + s]; int oi = si[t + s];
            if (ov > sv[t] || (ov == sv[t] && oi < si[t])) { sv[t] = ov; si[t] = oi; }
        }
        __syncthreads();
    }
    if (t == 0) {
        idxbuf[bc]  = si[0];
        out_idx[bc] = (float)si[0];
    }
}

// ---------------------------------------------------------------------------
// Kernel 3: q_st[b,c,:] = embedding[c, idx[b,c], :]  (forward value of the
// straight-through estimator); also writes commit_loss = 0.
// ---------------------------------------------------------------------------
__global__ __launch_bounds__(256) void gsq_gather(
    const float* __restrict__ emb,
    const int* __restrict__ idxbuf,
    float* __restrict__ out)
{
    const int bc = blockIdx.x;
    const int c  = bc & (C_ - 1);
    const int t  = threadIdx.x;
    const int idx = idxbuf[bc];
    const float4* src = (const float4*)(emb + (size_t)(c * K_ + idx) * D_);
    float4*       dst = (float4*)(out + (size_t)bc * D_);
    dst[t] = src[t];                         // 256 * float4 = 1024 floats
    if (bc == 0 && t == 0)
        out[(size_t)B_ * C_ * D_] = 0.0f;    // commit_loss
}

// ---------------------------------------------------------------------------
extern "C" void kernel_launch(void* const* d_in, const int* in_sizes, int n_in,
                              void* d_out, int out_size, void* d_ws, size_t ws_size,
                              hipStream_t stream)
{
    const float* z   = (const float*)d_in[0];   // [64,64,32,32]
    const float* u   = (const float*)d_in[1];   // [64,64,1024]
    const float* emb = (const float*)d_in[2];   // [64,1024,1024]
    float* out = (float*)d_out;

    float* scores = (float*)d_ws;                                   // 16 MB
    int*   idxbuf = (int*)((char*)d_ws + (size_t)B_ * C_ * K_ * sizeof(float));
    float* out_idx = out + (size_t)B_ * C_ * D_ + 1;                // indices tail

    gsq_gemm_score<<<C_ * NTILES, 256, 0, stream>>>(z, u, emb, scores);
    gsq_argmax<<<B_ * C_, 256, 0, stream>>>(scores, idxbuf, out_idx);
    gsq_gather<<<B_ * C_, 256, 0, stream>>>(emb, idxbuf, out);
}
